// BahdanauAttention_56710748176622
// MI455X (gfx1250) — compile-verified, matrix operand order checked by reading, not run on hardware
//
#include <hip/hip_runtime.h>
#include <hip/hip_bf16.h>

typedef __attribute__((ext_vector_type(16))) _Float16 v16h;
typedef __attribute__((ext_vector_type(8)))  float    v8f;
typedef int b128_t __attribute__((vector_size(16)));   // pointee type for async-LDS builtin

#define BB 8
#define TQ 256
#define TV 512
#define DD 256
#define CTX_TOTAL ((long)BB * TQ * 2 * DD)   // floats in context region of d_out

#define QSTRIDE 257   // qs row stride (floats): broadcast reads, no align need
#define VSTRIDE 260   // vs row stride (floats): 16B-aligned rows for async B128

#define AS1 __attribute__((address_space(1)))
#define AS3 __attribute__((address_space(3)))

// ---- fast tanh: gfx1250 V_TANH_F32 (confirmed present) ----
#if defined(__has_builtin) && __has_builtin(__builtin_amdgcn_tanhf)
# define FAST_TANH(x) __builtin_amdgcn_tanhf(x)
#else
# define FAST_TANH(x) tanhf(x)
#endif

// ---- async global->LDS (ASYNCcnt) with sync fallback ----
#if defined(__has_builtin)
# if __has_builtin(__builtin_amdgcn_global_load_async_to_lds_b128)
#  define HAVE_ASYNC_LDS 1
# endif
#endif
#ifndef HAVE_ASYNC_LDS
# define HAVE_ASYNC_LDS 0
# pragma message("CDNA5 async path: fallback synchronous LDS copy")
#endif

#if HAVE_ASYNC_LDS
# if __has_builtin(__builtin_amdgcn_s_wait_asynccnt)
#  define WAIT_ASYNC() __builtin_amdgcn_s_wait_asynccnt(0)
# else
#  define WAIT_ASYNC() asm volatile("s_wait_asynccnt 0x0" ::: "memory")
# endif
#else
# define WAIT_ASYNC() ((void)0)
#endif

// Copy one 32 x 256 f32 tile (row-major, src stride DD) into LDS with VSTRIDE pad.
// 2048 16B chunks; 8 per thread; consecutive lanes -> consecutive 16B (coalesced).
__device__ __forceinline__ void issue_vtile(const float* src, float* dst, int t) {
#if HAVE_ASYNC_LDS
#pragma unroll
    for (int i = 0; i < 8; ++i) {
        int k = t + i * 256;
        int r = k >> 6;
        int c = (k & 63) << 2;
        __builtin_amdgcn_global_load_async_to_lds_b128(
            (AS1 b128_t*)(src + r * DD + c),
            (AS3 b128_t*)(dst + r * VSTRIDE + c), 0, 0);
    }
#else
    for (int idx = t; idx < 32 * DD; idx += 256)
        dst[(idx >> 8) * VSTRIDE + (idx & 255)] = src[idx];
#endif
}

// K-element index for 16-bit A/B matrices (16x32 / 32x16), per ISA layout:
// lanes 0-15 hold K = 0..7,16..23 ; lanes 16-31 hold K = 8..15,24..31
__device__ __forceinline__ int wmma_k_of(int e, int half) {
    return half * 8 + e + ((e >= 8) ? 8 : 0);
}

// ---------------------------------------------------------------------------
// Projection: out[b] (T x 256) = in[b] (T x 256) @ W (256 x 256) + bias
// One wave computes one 16x16 output tile via 8x v_wmma_f32_16x16x32_f16.
// ---------------------------------------------------------------------------
__global__ __launch_bounds__(256) void proj_kernel(
    const float* __restrict__ in, const float* __restrict__ W,
    const float* __restrict__ bias, float* __restrict__ out, int T) {
    const int wave = threadIdx.x >> 5;
    const int lane = threadIdx.x & 31;
    const int half = lane >> 4;
    const int mn   = lane & 15;

    const int ntiles = DD / 16;
    const int tilesPerBatch = (T / 16) * ntiles;
    int tile = blockIdx.x * (blockDim.x >> 5) + wave;
    int b  = tile / tilesPerBatch;
    int r  = tile % tilesPerBatch;
    int m0 = (r / ntiles) * 16;
    int n0 = (r % ntiles) * 16;
    if (b >= BB) return;

    const float* inB  = in  + (long)b * T * DD;
    float*       outB = out + (long)b * T * DD;

    v8f c = {};
    for (int k0 = 0; k0 < DD; k0 += 32) {
        const float4* ap = (const float4*)(inB + (m0 + mn) * DD + k0 + half * 8);
        float4 a0 = ap[0], a1 = ap[1];   // K offsets 0..7
        float4 a2 = ap[4], a3 = ap[5];   // K offsets 16..23
        v16h a;
        a[0]=(_Float16)a0.x; a[1]=(_Float16)a0.y; a[2]=(_Float16)a0.z; a[3]=(_Float16)a0.w;
        a[4]=(_Float16)a1.x; a[5]=(_Float16)a1.y; a[6]=(_Float16)a1.z; a[7]=(_Float16)a1.w;
        a[8]=(_Float16)a2.x; a[9]=(_Float16)a2.y; a[10]=(_Float16)a2.z; a[11]=(_Float16)a2.w;
        a[12]=(_Float16)a3.x; a[13]=(_Float16)a3.y; a[14]=(_Float16)a3.z; a[15]=(_Float16)a3.w;

        v16h bm;
#pragma unroll
        for (int e = 0; e < 16; ++e) {
            int K = k0 + wmma_k_of(e, half);
            bm[e] = (_Float16)W[K * DD + (n0 + mn)];
        }
        c = __builtin_amdgcn_wmma_f32_16x16x32_f16(false, a, false, bm,
                                                   (short)0, c, false, false);
    }
    float bv = bias[n0 + mn];
#pragma unroll
    for (int e = 0; e < 8; ++e) {
        int M = e + half * 8;
        outB[(m0 + M) * DD + (n0 + mn)] = c[e] + bv;
    }
}

// ---------------------------------------------------------------------------
// Score: raw score[b,q,v] = sum_d tanh(qp[b,q,d] + vp[b,v,d]) -> alignment_t
// region of d_out ([B][TV][TQ]). Double-buffered async v-tiles; query concat
// copy piggy-backed. Dynamic LDS: qs 16xQSTRIDE + 2 x (32xVSTRIDE).
// ---------------------------------------------------------------------------
__global__ __launch_bounds__(256) void score_kernel(
    const float* __restrict__ qp, const float* __restrict__ vp,
    const float* __restrict__ query, float* __restrict__ out) {
    extern __shared__ float smem[];
    float* qs  = smem;                        // [16][QSTRIDE]
    float* vsA = smem + 16 * QSTRIDE;         // [32][VSTRIDE]
    float* vsB = vsA + 32 * VSTRIDE;          // [32][VSTRIDE]

    const int b  = blockIdx.x >> 4;           // TQ/16 q-tiles per batch
    const int q0 = (blockIdx.x & 15) * 16;
    const int t  = threadIdx.x;
    const int w    = t >> 5;                  // wave id 0..7 -> rows {w, w+8}
    const int lane = t & 31;                  // j within 32-wide v tile

    const float* qpB = qp + ((long)b * TQ + q0) * DD;
    for (int idx = t; idx < 16 * DD; idx += 256)
        qs[(idx >> 8) * QSTRIDE + (idx & 255)] = qpB[idx];

    // copy query into concat half of context output: out[b][q][256+d]
    const float* quB = query + ((long)b * TQ + q0) * DD;
    float* outc = out + ((long)b * TQ + q0) * (2 * DD);
    for (int idx = t; idx < 16 * DD; idx += 256)
        outc[(idx >> 8) * 2 * DD + DD + (idx & 255)] = quB[idx];

    float* outA = out + CTX_TOTAL + (long)b * TV * TQ;   // [TV][TQ]
    const float* vBase = vp + (long)b * TV * DD;

    // prologue: fetch tile 0
    issue_vtile(vBase, vsA, t);
    WAIT_ASYNC();
    __syncthreads();

    for (int jt = 0; jt < TV / 32; ++jt) {
        float* cur = (jt & 1) ? vsB : vsA;
        float* nxt = (jt & 1) ? vsA : vsB;
        if (jt + 1 < TV / 32)
            issue_vtile(vBase + (long)(jt + 1) * 32 * DD, nxt, t);

        const float* vrow = cur + lane * VSTRIDE;
        const float* qr0  = qs + w * QSTRIDE;
        const float* qr1  = qs + (w + 8) * QSTRIDE;
        float s0 = 0.f, s1 = 0.f;
#pragma unroll 4
        for (int d = 0; d < DD; ++d) {
            float va = vrow[d];
            s0 += FAST_TANH(qr0[d] + va);
            s1 += FAST_TANH(qr1[d] + va);
        }
        int v = jt * 32 + lane;
        outA[(long)v * TQ + q0 + w]     = s0;
        outA[(long)v * TQ + q0 + w + 8] = s1;

        WAIT_ASYNC();       // next tile landed (this wave's issues)
        __syncthreads();    // publish to all waves; protect recycled buffer
    }
}

// ---------------------------------------------------------------------------
// Softmax over v (stride-TQ columns) in-place on alignment_t region.
// ---------------------------------------------------------------------------
__global__ __launch_bounds__(256) void softmax_kernel(float* __restrict__ alignT) {
    __shared__ float red[16][17];
    const int b  = blockIdx.x >> 4;
    const int q0 = (blockIdx.x & 15) * 16;
    float* A = alignT + (long)b * TV * TQ;
    const int vv = threadIdx.x >> 4;
    const int qq = threadIdx.x & 15;
    const int q  = q0 + qq;

    float m = -3.0e38f;
    for (int v = vv; v < TV; v += 16) m = fmaxf(m, A[(long)v * TQ + q]);
    red[qq][vv] = m;
    __syncthreads();
    if (vv == 0) {
        float mm = red[qq][0];
        for (int k = 1; k < 16; ++k) mm = fmaxf(mm, red[qq][k]);
        red[qq][16] = mm;
    }
    __syncthreads();
    float rowmax = red[qq][16];

    float sum = 0.f;
    for (int v = vv; v < TV; v += 16) sum += __expf(A[(long)v * TQ + q] - rowmax);
    __syncthreads();
    red[qq][vv] = sum;
    __syncthreads();
    if (vv == 0) {
        float ss = 0.f;
        for (int k = 0; k < 16; ++k) ss += red[qq][k];
        red[qq][16] = ss;
    }
    __syncthreads();
    float inv = 1.0f / red[qq][16];
    for (int v = vv; v < TV; v += 16) {
        long idx = (long)v * TQ + q;
        A[idx] = __expf(A[idx] - rowmax) * inv;
    }
}

// ---------------------------------------------------------------------------
// Context: context[b] (TQ x DD) = alignment[b] (TQ x TV) @ value[b] (TV x DD)
// A(M,K) = alignT[K][M]; one wave per 16x16 tile, 16 WMMAs over K=512.
// ---------------------------------------------------------------------------
__global__ __launch_bounds__(256) void context_kernel(
    const float* __restrict__ value, const float* __restrict__ alignT,
    float* __restrict__ out) {
    const int wave = threadIdx.x >> 5;
    const int lane = threadIdx.x & 31;
    const int half = lane >> 4;
    const int mn   = lane & 15;

    const int ntiles = DD / 16;                 // 16
    const int perB   = (TQ / 16) * ntiles;      // 256
    int tile = blockIdx.x * (blockDim.x >> 5) + wave;
    int b  = tile / perB;
    int r  = tile % perB;
    int m0 = (r / ntiles) * 16;
    int n0 = (r % ntiles) * 16;
    if (b >= BB) return;

    const float* alB = alignT + (long)b * TV * TQ;   // [TV][TQ]
    const float* vaB = value  + (long)b * TV * DD;   // [TV][DD]

    v8f c = {};
    for (int k0 = 0; k0 < TV; k0 += 32) {
        v16h a, bm;
#pragma unroll
        for (int e = 0; e < 16; ++e) {
            int K = k0 + wmma_k_of(e, half);
            a[e]  = (_Float16)alB[(long)K * TQ + (m0 + mn)];
            bm[e] = (_Float16)vaB[K * DD + (n0 + mn)];
        }
        c = __builtin_amdgcn_wmma_f32_16x16x32_f16(false, a, false, bm,
                                                   (short)0, c, false, false);
    }
#pragma unroll
    for (int e = 0; e < 8; ++e) {
        int M = e + half * 8;
        out[((long)b * TQ + m0 + M) * (2 * DD) + (n0 + mn)] = c[e];
    }
}

extern "C" void kernel_launch(void* const* d_in, const int* in_sizes, int n_in,
                              void* d_out, int out_size, void* d_ws, size_t ws_size,
                              hipStream_t stream) {
    const float* query = (const float*)d_in[0];   // [8,256,256]
    const float* value = (const float*)d_in[1];   // [8,512,256]
    const float* W1    = (const float*)d_in[2];   // [256,256]
    const float* b1    = (const float*)d_in[3];   // [256]
    const float* W2    = (const float*)d_in[4];   // [256,256]
    const float* b2    = (const float*)d_in[5];   // [256]
    float* out = (float*)d_out;

    float* qp = (float*)d_ws;                     // [8,256,256] = 2 MB
    float* vp = qp + (long)BB * TQ * DD;          // [8,512,256] = 4 MB

    proj_kernel<<<256, 256, 0, stream>>>(query, W1, b1, qp, TQ);
    proj_kernel<<<512, 256, 0, stream>>>(value, W2, b2, vp, TV);

    // dynamic LDS: 16*QSTRIDE + 2*32*VSTRIDE floats = 83,008 bytes
    const size_t smem_bytes = (size_t)(16 * QSTRIDE + 2 * 32 * VSTRIDE) * sizeof(float);
    score_kernel<<<BB * (TQ / 16), 256, smem_bytes, stream>>>(qp, vp, query, out);

    float* alignT = out + CTX_TOTAL;
    softmax_kernel<<<BB * (TQ / 16), 256, 0, stream>>>(alignT);

    context_kernel<<<256, 256, 0, stream>>>(value, alignT, out);
}
